// SolidityGNN_38500086841935
// MI455X (gfx1250) — compile-verified, matrix operand order checked by reading, not run on hardware
//
#include <hip/hip_runtime.h>
#include <hip/hip_bf16.h>

// ---------------------------------------------------------------------------
// SolidityGNN (2-layer GraphSAGE + global mean pool) for MI455X / gfx1250.
//
// Phases (all on `stream`, workspace in d_ws):
//   1. zero agg/deg
//   2. edge scatter: agg[dst] += x[src], deg[dst] += 1   (fp32 global atomics)
//   3. conv1: h1 = relu((agg/deg) @ WL1 + bl1 + x @ WR1)  (bf16 WMMA, f32 acc)
//   4. zero agg/deg
//   5. edge scatter with h1
//   6. conv2: h2 = relu((agg/deg) @ WL2 + bl2 + h1 @ WR2)
//   7. zero pooled/cnt ; pool scatter (atomics) ; tiny head GEMM -> d_out
// ---------------------------------------------------------------------------

#define N_NODES 40000
#define N_EDGES 640000
#define CH      128
#define NGRAPH  64

typedef __attribute__((ext_vector_type(16))) __bf16 v16bf;
typedef __attribute__((ext_vector_type(8)))  __bf16 v8bf;
typedef __attribute__((ext_vector_type(8)))  float  v8f;

__device__ __forceinline__ __bf16 f2bf(float f) {
  union { float f; unsigned u; } a; a.f = f;
  unsigned u = a.u + 0x7FFFu + ((a.u >> 16) & 1u);   // round-to-nearest-even
  union { unsigned short s; __bf16 b; } r;
  r.s = (unsigned short)(u >> 16);
  return r.b;
}

__device__ __forceinline__ void atomAddF(float* p, float v) {
  (void)__hip_atomic_fetch_add(p, v, __ATOMIC_RELAXED, __HIP_MEMORY_SCOPE_AGENT);
}

// ---------------------------------------------------------------- zero ------
__global__ __launch_bounds__(256) void zero_kernel(float4* __restrict__ p, int n4) {
  int i = blockIdx.x * blockDim.x + threadIdx.x;
  int stride = gridDim.x * blockDim.x;
  float4 z = make_float4(0.f, 0.f, 0.f, 0.f);
  for (; i < n4; i += stride) p[i] = z;
}

// ------------------------------------------------------------- scatter ------
// One wave32 per edge: lane l adds 4 consecutive floats of feat[src] into
// agg[dst]; lane 0 bumps deg[dst].
__global__ __launch_bounds__(256) void scatter_kernel(
    const float* __restrict__ feat, const int* __restrict__ ei,
    float* __restrict__ agg, float* __restrict__ deg) {
  const int lane = threadIdx.x & 31;
  int wid = (blockIdx.x * blockDim.x + threadIdx.x) >> 5;
  const int nw = (gridDim.x * blockDim.x) >> 5;
  for (int e = wid; e < N_EDGES; e += nw) {
    const unsigned src = (unsigned)ei[e];
    const unsigned dst = (unsigned)ei[N_EDGES + e];
    float4 v = reinterpret_cast<const float4*>(feat + (size_t)src * CH)[lane];
    float* a = agg + (size_t)dst * CH + lane * 4;
    atomAddF(a + 0, v.x);
    atomAddF(a + 1, v.y);
    atomAddF(a + 2, v.z);
    atomAddF(a + 3, v.w);
    if (lane == 0) atomAddF(deg + dst, 1.0f);
  }
}

// ---------------------------------------------------------------- conv ------
// out = relu((agg/max(deg,1)) @ WL + BL + xin @ WR), all [N,128]@[128,128].
// Block = 8 wave32; wave w owns output columns [16w, 16w+16).
// Weight B-fragments are register-resident per wave, reused over a
// grid-stride loop of 16-node M-tiles. A-tiles staged through LDS as bf16.
// All 8 A-fragments are loaded ahead of the WMMA chain so the ds_loads can
// be claused and overlapped with WMMA issue (partial s_wait_dscnt counts).
__global__ __launch_bounds__(256) void conv_kernel(
    const float* __restrict__ xin, const float* __restrict__ agg,
    const float* __restrict__ deg,
    const float* __restrict__ WL, const float* __restrict__ BL,
    const float* __restrict__ WR,
    float* __restrict__ out) {
  const int lane = threadIdx.x & 31;
  const int wave = threadIdx.x >> 5;      // 0..7
  const int n0   = wave * 16;
  const int half = lane >> 4;             // K half-select within 32-K fragment
  const int col  = lane & 15;             // output column within tile

  // ---- load weight fragments once (WMMA B layout: lane = N, pairs of K) ----
  v16bf bWL[4], bWR[4];
#pragma unroll
  for (int kk = 0; kk < 4; ++kk) {
    v16bf fl, fr;
#pragma unroll
    for (int j = 0; j < 16; ++j) {
      int k = kk * 32 + (j < 8 ? half * 8 + j : 16 + half * 8 + (j - 8));
      fl[j] = f2bf(WL[k * CH + n0 + col]);
      fr[j] = f2bf(WR[k * CH + n0 + col]);
    }
    bWL[kk] = fl;
    bWR[kk] = fr;
  }
  const float bias = BL[n0 + col];

  // 136-element row pitch (272B = 17*16B) -> conflict-free 16B frag reads
  __shared__ __bf16 sAgg[16 * 136];
  __shared__ __bf16 sX[16 * 136];

  const int srow = threadIdx.x >> 4;        // staging row 0..15
  const int scb  = (threadIdx.x & 15) * 8;  // staging col base 0..120
  const int mstep = gridDim.x * 16;         // nodes per grid-stride step
  const size_t pstep = (size_t)mstep * CH;  // float elements per step

  // incremental per-thread pointers (advance by pstep each iteration)
  const int m0_0 = blockIdx.x * 16;
  const float* ap = agg + (size_t)(m0_0 + srow) * CH + scb;
  const float* xp = xin + (size_t)(m0_0 + srow) * CH + scb;
  const float* dp = deg + m0_0 + srow;
  float*       ob = out + (size_t)(m0_0 + half * 8) * CH + n0 + col;

  __bf16* sA  = &sAgg[srow * 136 + scb];
  __bf16* sXp = &sX[srow * 136 + scb];
  const int arow = lane & 15;

  for (int m0 = m0_0; m0 < N_NODES; m0 += mstep) {
    // ---- stage A tiles (f32 -> bf16, deg-normalize the aggregate) ----
    const float d   = *dp;
    const float inv = 1.0f / fmaxf(d, 1.0f);
    float4 a0 = reinterpret_cast<const float4*>(ap)[0];
    float4 a1 = reinterpret_cast<const float4*>(ap)[1];
    float4 x0 = reinterpret_cast<const float4*>(xp)[0];
    float4 x1 = reinterpret_cast<const float4*>(xp)[1];
    v8bf va, vx;
    va[0] = f2bf(a0.x * inv); va[1] = f2bf(a0.y * inv);
    va[2] = f2bf(a0.z * inv); va[3] = f2bf(a0.w * inv);
    va[4] = f2bf(a1.x * inv); va[5] = f2bf(a1.y * inv);
    va[6] = f2bf(a1.z * inv); va[7] = f2bf(a1.w * inv);
    vx[0] = f2bf(x0.x); vx[1] = f2bf(x0.y); vx[2] = f2bf(x0.z); vx[3] = f2bf(x0.w);
    vx[4] = f2bf(x1.x); vx[5] = f2bf(x1.y); vx[6] = f2bf(x1.z); vx[7] = f2bf(x1.w);
    *reinterpret_cast<v8bf*>(sA)  = va;
    *reinterpret_cast<v8bf*>(sXp) = vx;

    // prefetch next tile (uniform branch; emits global_prefetch_b8)
    if (m0 + mstep < N_NODES) {
      __builtin_prefetch(ap + pstep, 0, 1);
      __builtin_prefetch(xp + pstep, 0, 1);
    }
    __syncthreads();

    // ---- load all 8 A-fragments first (claused ds_load_b128, deep DScnt) --
    v16bf aFrag[8];
#pragma unroll
    for (int kk = 0; kk < 4; ++kk) {
      v8bf lo = *reinterpret_cast<const v8bf*>(&sAgg[arow * 136 + kk * 32 + half * 8]);
      v8bf hi = *reinterpret_cast<const v8bf*>(&sAgg[arow * 136 + kk * 32 + 16 + half * 8]);
      aFrag[kk] = __builtin_shufflevector(lo, hi, 0, 1, 2, 3, 4, 5, 6, 7,
                                          8, 9, 10, 11, 12, 13, 14, 15);
    }
#pragma unroll
    for (int kk = 0; kk < 4; ++kk) {
      v8bf lo = *reinterpret_cast<const v8bf*>(&sX[arow * 136 + kk * 32 + half * 8]);
      v8bf hi = *reinterpret_cast<const v8bf*>(&sX[arow * 136 + kk * 32 + 16 + half * 8]);
      aFrag[4 + kk] = __builtin_shufflevector(lo, hi, 0, 1, 2, 3, 4, 5, 6, 7,
                                              8, 9, 10, 11, 12, 13, 14, 15);
    }

    // ---- 8 back-to-back WMMAs into one f32 accumulator ----
    v8f c = {0.f, 0.f, 0.f, 0.f, 0.f, 0.f, 0.f, 0.f};
#pragma unroll
    for (int kk = 0; kk < 4; ++kk)
      c = __builtin_amdgcn_wmma_f32_16x16x32_bf16(false, aFrag[kk], false, bWL[kk],
                                                  (short)0, c, false, false);
#pragma unroll
    for (int kk = 0; kk < 4; ++kk)
      c = __builtin_amdgcn_wmma_f32_16x16x32_bf16(false, aFrag[4 + kk], false, bWR[kk],
                                                  (short)0, c, false, false);

    // ---- bias + relu epilogue; constant offsets i*CH -> s_clause stores ----
#pragma unroll
    for (int i = 0; i < 8; ++i) {
      float v = fmaxf(c[i] + bias, 0.0f);
      ob[(size_t)i * CH] = v;
    }
    __syncthreads();

    ap += pstep;
    xp += pstep;
    dp += mstep;
    ob += pstep;
  }
}

// ---------------------------------------------------------------- pool ------
__global__ __launch_bounds__(256) void pool_kernel(
    const float* __restrict__ h, const int* __restrict__ batch,
    float* __restrict__ pooled, float* __restrict__ cnt) {
  const int lane = threadIdx.x & 31;
  int wid = (blockIdx.x * blockDim.x + threadIdx.x) >> 5;
  const int nw = (gridDim.x * blockDim.x) >> 5;
  for (int n = wid; n < N_NODES; n += nw) {
    const unsigned g = (unsigned)batch[n];
    float4 v = reinterpret_cast<const float4*>(h + (size_t)n * CH)[lane];
    float* p = pooled + (size_t)g * CH + lane * 4;
    atomAddF(p + 0, v.x);
    atomAddF(p + 1, v.y);
    atomAddF(p + 2, v.z);
    atomAddF(p + 3, v.w);
    if (lane == 0) atomAddF(cnt + g, 1.0f);
  }
}

// ---------------------------------------------------------------- head ------
__global__ __launch_bounds__(128) void head_kernel(
    const float* __restrict__ pooled, const float* __restrict__ cnt,
    const float* __restrict__ wlin, const float* __restrict__ blin,
    float* __restrict__ out) {
  const int t = threadIdx.x;
  if (t >= NGRAPH * 2) return;
  const int g = t >> 1, o = t & 1;
  const float inv = 1.0f / fmaxf(cnt[g], 1.0f);
  float acc = blin[o];
#pragma unroll 8
  for (int k = 0; k < CH; ++k)
    acc = fmaf(pooled[g * CH + k] * inv, wlin[k * 2 + o], acc);
  out[g * 2 + o] = acc;
}

// --------------------------------------------------------------- launch -----
extern "C" void kernel_launch(void* const* d_in, const int* in_sizes, int n_in,
                              void* d_out, int out_size, void* d_ws, size_t ws_size,
                              hipStream_t stream) {
  const float* x     = (const float*)d_in[0];
  const int*   ei    = (const int*)d_in[1];   // [2, E] int32 (harness contract)
  const int*   batch = (const int*)d_in[2];   // [N] int32
  const float* wl1   = (const float*)d_in[3];
  const float* bl1   = (const float*)d_in[4];
  const float* wr1   = (const float*)d_in[5];
  const float* wl2   = (const float*)d_in[6];
  const float* bl2   = (const float*)d_in[7];
  const float* wr2   = (const float*)d_in[8];
  const float* wlin  = (const float*)d_in[9];
  const float* blin  = (const float*)d_in[10];
  float* out = (float*)d_out;

  // workspace layout (floats)
  float* agg    = (float*)d_ws;                 // N*CH
  float* deg    = agg + (size_t)N_NODES * CH;   // N
  float* h1     = deg + N_NODES;                // N*CH
  float* h2     = h1 + (size_t)N_NODES * CH;    // N*CH
  float* pooled = h2 + (size_t)N_NODES * CH;    // NGRAPH*CH
  float* cnt    = pooled + NGRAPH * CH;         // NGRAPH

  const int aggdeg4 = (N_NODES * CH + N_NODES) / 4;
  const int pool4   = (NGRAPH * CH + NGRAPH) / 4;

  // ---- conv 1 ----
  zero_kernel<<<1024, 256, 0, stream>>>((float4*)agg, aggdeg4);
  scatter_kernel<<<2000, 256, 0, stream>>>(x, ei, agg, deg);
  conv_kernel<<<640, 256, 0, stream>>>(x, agg, deg, wl1, bl1, wr1, h1);

  // ---- conv 2 ----
  zero_kernel<<<1024, 256, 0, stream>>>((float4*)agg, aggdeg4);
  scatter_kernel<<<2000, 256, 0, stream>>>(h1, ei, agg, deg);
  conv_kernel<<<640, 256, 0, stream>>>(h1, agg, deg, wl2, bl2, wr2, h2);

  // ---- pool + head ----
  zero_kernel<<<16, 256, 0, stream>>>((float4*)pooled, pool4);
  pool_kernel<<<1250, 256, 0, stream>>>(h2, batch, pooled, cnt);
  head_kernel<<<1, 128, 0, stream>>>(pooled, cnt, wlin, blin, out);
}
